// ScaledDotProductAttention1_17377437679899
// MI455X (gfx1250) — matrix operand order repaired and checked by reading, not compiled, not run on hardware
//
#include <hip/hip_runtime.h>
#include <hip/hip_bf16.h>
#include <stdint.h>

typedef _Float16 half16 __attribute__((ext_vector_type(16)));
typedef float    float8 __attribute__((ext_vector_type(8)));

// ---------------------------------------------------------------------------
// WMMA helpers (CDNA5: V_WMMA_F32_16X16X32_F16, wave32)
// ---------------------------------------------------------------------------
__device__ __forceinline__ float8 wmma32(half16 a, half16 b, float8 c) {
    return __builtin_amdgcn_wmma_f32_16x16x32_f16(
        /*neg_a=*/false, a, /*neg_b=*/false, b,
        /*c_mod=*/(short)0, c, /*reuse_a=*/false, /*reuse_b=*/false);
}

// A-matrix fragment, 16x32 f16, from row-major [M][K] LDS (ISA 7.12.2 layout):
// lanes 0-15 / 16-31 both map M = row0 + (lane&15); halves 0..7 -> K = hi*8+0..7,
// halves 8..15 -> K = 16 + hi*8 + 0..7 (hi = lane>>4).
__device__ __forceinline__ half16 ldsFragA(const _Float16* __restrict__ base,
                                           int row0, int stride, int kbase, int kmax) {
    const int lane = threadIdx.x & 31;
    const int r  = row0 + (lane & 15);
    const int k0 = kbase + ((lane >> 4) << 3);
    const _Float16* p = base + r * stride;
    half16 f;
#pragma unroll
    for (int i = 0; i < 8; ++i) {
        int k = k0 + i;          int kc = (k < kmax) ? k : 0;
        _Float16 t = p[kc];      f[i] = (k < kmax) ? t : (_Float16)0;
    }
#pragma unroll
    for (int i = 0; i < 8; ++i) {
        int k = k0 + 16 + i;     int kc = (k < kmax) ? k : 0;
        _Float16 t = p[kc];      f[8 + i] = (k < kmax) ? t : (_Float16)0;
    }
    return f;
}

// A fragment with row shift + zero pad (for the 3x3 conv taps). kmax fixed 64.
__device__ __forceinline__ half16 ldsFragAShift(const _Float16* __restrict__ base,
                                                int row0, int dl, int stride, int kbase) {
    const int lane = threadIdx.x & 31;
    int r = row0 + (lane & 15) + dl;
    const bool ok = (r >= 0) && (r < 144);
    const int rc = ok ? r : 0;
    const int k0 = kbase + ((lane >> 4) << 3);
    const _Float16* p = base + rc * stride;
    half16 f;
#pragma unroll
    for (int i = 0; i < 8; ++i)  { _Float16 t = p[k0 + i];      f[i]     = ok ? t : (_Float16)0; }
#pragma unroll
    for (int i = 0; i < 8; ++i)  { _Float16 t = p[k0 + 16 + i]; f[8 + i] = ok ? t : (_Float16)0; }
    return f;
}

// B-matrix fragment, 32x16 f16, from row-major [N][K] LDS:
// lane -> N = row0 + (lane&15); halves 0..7 -> K = kbase + hi*8 + 0..7,
// halves 8..15 -> K = kbase + 16 + hi*8 + 0..7.
__device__ __forceinline__ half16 ldsFragB(const _Float16* __restrict__ base,
                                           int row0, int stride, int kbase, int kmax) {
    const int lane = threadIdx.x & 31;
    const int r  = row0 + (lane & 15);
    const int k0 = kbase + ((lane >> 4) << 3);
    const _Float16* p = base + r * stride;
    half16 f;
#pragma unroll
    for (int i = 0; i < 16; ++i) {
        int k = k0 + i + ((i >> 3) << 3); int kc = (k < kmax) ? k : 0;
        _Float16 t = p[kc];      f[i] = (k < kmax) ? t : (_Float16)0;
    }
    return f;
}

// ---------------------------------------------------------------------------
// Tensor Data Mover: 1-D f16 copy (global -> LDS). D# per ISA 08_async_tensor §8.
// ---------------------------------------------------------------------------
#if defined(__HIP_DEVICE_COMPILE__) && __has_builtin(__builtin_amdgcn_tensor_load_to_lds)
#define HAVE_TDM 1
#else
#define HAVE_TDM 0
#endif

#if HAVE_TDM
typedef unsigned tdm_u4 __attribute__((ext_vector_type(4)));
typedef int      tdm_i4 __attribute__((ext_vector_type(4)));
typedef int      tdm_i8 __attribute__((ext_vector_type(8)));

// nElem = number of 2-byte elements, must be <= 65535 (16-bit tile_dim0)
__device__ __forceinline__ void tdm_copy_f16_1d(void* ldsDst, const void* gSrc, unsigned nElem) {
    unsigned long long ga = (unsigned long long)(uintptr_t)gSrc;
    unsigned ldsOff = (unsigned)(uintptr_t)ldsDst;   // LDS aperture: low 32 bits = LDS addr
    tdm_u4 g0;
    g0[0] = 1u;                                       // count=1, user mode, no gather
    g0[1] = ldsOff;                                   // lds_addr
    g0[2] = (unsigned)ga;                             // global_addr[31:0]
    g0[3] = (unsigned)((ga >> 32) & 0x01FFFFFFu) | 0x80000000u; // ga[56:32] | type=2
    tdm_i8 g1;
    g1[0] = 0x00010000;                               // wg_mask=0, data_size=1 (2 bytes)
    g1[1] = (int)((nElem & 0xFFFFu) << 16);           // tensor_dim0[15:0]
    g1[2] = (int)(((nElem >> 16) & 0xFFFFu) | (1u << 16)); // tensor_dim0[31:16], tensor_dim1=1
    g1[3] = (int)((nElem & 0xFFFFu) << 16);           // tile_dim0 = nElem
    g1[4] = 1;                                        // tile_dim1 = 1, tile_dim2 = 0
    g1[5] = (int)nElem;                               // tensor_dim0_stride[31:0]
    g1[6] = 0;
    g1[7] = 0;
    tdm_i4 z4 = {0, 0, 0, 0};
#if defined(__clang_major__) && (__clang_major__ >= 23)
    tdm_i8 z8 = {0, 0, 0, 0, 0, 0, 0, 0};
    __builtin_amdgcn_tensor_load_to_lds(g0, g1, z4, z4, z8, 0);
#else
    __builtin_amdgcn_tensor_load_to_lds(g0, g1, z4, z4, 0);
#endif
}
#endif

// ---------------------------------------------------------------------------
// Prep: ppT diagonal table (287 entries) + f16 weight repack into workspace
// ws layout: [0)  ppTd      287 f32
//            [2048) dwH     [9][64][64] f16  (tap, o, i)
//            [75776) pwH    [64][64] f16     (o, i)
// ---------------------------------------------------------------------------
__global__ void prep_kernel(const float* __restrict__ dw_w, const float* __restrict__ pw_w,
                            float* __restrict__ ppTd, _Float16* __restrict__ dwH,
                            _Float16* __restrict__ pwH) {
    if (blockIdx.x == 160) {
        int t = threadIdx.x;
        if (t < 287) {
            float delta = (float)(t - 143);
            float s = 0.f;
            for (int j = 0; j < 32; ++j) {
                float w = __expf(-(float)j * 0.28782313f); // 10000^(-j/32)
                s += cosf(delta * w);
            }
            ppTd[t] = s;
        }
        return;
    }
    int gid = blockIdx.x * 256 + threadIdx.x;
    if (gid < 36864) {
        int tap = gid >> 12, rem = gid & 4095;
        int o = rem >> 6, i = rem & 63;
        dwH[gid] = (_Float16)dw_w[o * 576 + i * 9 + tap];
    } else if (gid < 40960) {
        int g2 = gid - 36864;
        pwH[g2] = (_Float16)pw_w[g2];
    }
}

// ---------------------------------------------------------------------------
// Attention kernel: one workgroup (256 thr = 8 waves) per (b,h)
// LDS: S fp32 [144][144] | (qh,kh f16 phase1) overlaid by (attnH f16, vT f16) | ppTd
// ---------------------------------------------------------------------------
#define LA_Q    82944
#define LA_K    101376
#define LA_ATTN 82944
#define LA_VT   124416
#define LA_PPT  144896
#define LA_SIZE 146048

__global__ __launch_bounds__(256) void attn_kernel(const float* __restrict__ q,
                                                   const float* __restrict__ k,
                                                   const float* __restrict__ v,
                                                   const float* __restrict__ ppTd_g,
                                                   const float* __restrict__ biasPtr,
                                                   float* __restrict__ ctx,
                                                   float* __restrict__ attnOut) {
    __shared__ __align__(16) unsigned char smem[LA_SIZE];
    float*     S     = (float*)smem;
    _Float16*  qh    = (_Float16*)(smem + LA_Q);
    _Float16*  kh    = (_Float16*)(smem + LA_K);
    _Float16*  attnH = (_Float16*)(smem + LA_ATTN);
    _Float16*  vT    = (_Float16*)(smem + LA_VT);
    float*     ppTd  = (float*)(smem + LA_PPT);

    const int bh = blockIdx.x;
    const int tid = threadIdx.x;
    const int lane = tid & 31, wid = tid >> 5;
    const size_t base = (size_t)bh * 9216;

    for (int i = tid; i < 287; i += 256) ppTd[i] = ppTd_g[i];
    for (int i = tid; i < 9216; i += 256) {
        int l = i >> 6, d = i & 63;
        qh[i] = (_Float16)(q[base + i] * 0.125f);   // fold 1/sqrt(D)
        kh[i] = (_Float16)(k[base + i]);
        vT[d * 160 + l] = (_Float16)(v[base + i]);  // transposed for B-frags
    }
    __syncthreads();

    // ---- S = (q/8) @ k^T  (positional bias folded in during phase 3) -------
    for (int t = wid; t < 81; t += 8) {
        int mt = t / 9, nt = t % 9;
        float8 acc = {};
        acc = wmma32(ldsFragA(qh, mt * 16, 64, 0, 64),  ldsFragB(kh, nt * 16, 64, 0, 64),  acc);
        acc = wmma32(ldsFragA(qh, mt * 16, 64, 32, 64), ldsFragB(kh, nt * 16, 64, 32, 64), acc);
        int n  = nt * 16 + (lane & 15);
        int mb = mt * 16 + ((lane >> 4) << 3);
#pragma unroll
        for (int r = 0; r < 8; ++r)
            S[(mb + r) * 144 + n] = acc[r];
    }
    __syncthreads();

    // ---- per-row top-K threshold (K=102) + masked softmax ------------------
    const float invBias = 1.0f / biasPtr[0];
    for (int row = wid; row < 144; row += 8) {      // one wave per row
        float sv[5]; unsigned uu[5];
#pragma unroll
        for (int e = 0; e < 5; ++e) {
            int j = lane + 32 * e;
            float x = 0.0f;
            if (j < 144) x = S[row * 144 + j] + ppTd[row - j + 143];
            sv[e] = x;
            unsigned u = __float_as_uint(x);
            u ^= ((unsigned)((int)u >> 31)) | 0x80000000u;  // order-preserving map
            uu[e] = (j < 144) ? u : 0u;
        }
        unsigned th = 0;                             // MSB-first binary search
        for (int bit = 31; bit >= 0; --bit) {
            unsigned cand = th | (1u << bit);
            int cnt = 0;
#pragma unroll
            for (int e = 0; e < 5; ++e)
                cnt += __popcll(__ballot(uu[e] >= cand));
            if (cnt >= 102) th = cand;
        }
        float ev[5]; float lsum = 0.f;
#pragma unroll
        for (int e = 0; e < 5; ++e) {
            float x = (uu[e] >= th) ? __expf(sv[e] * invBias) : 0.f;
            ev[e] = x; lsum += x;
        }
#pragma unroll
        for (int off = 16; off > 0; off >>= 1) lsum += __shfl_xor(lsum, off);
        float rinv = 1.0f / lsum;
#pragma unroll
        for (int e = 0; e < 5; ++e) {
            int j = lane + 32 * e;
            if (j < 144) {
                float a = ev[e] * rinv;
                attnOut[(size_t)bh * 20736 + row * 144 + j] = a;
                attnH[row * 144 + j] = (_Float16)a;
            }
        }
    }
    __syncthreads();

    // ---- context = attn @ v  (K=144, padded tiles zero-filled) -------------
    for (int t = wid; t < 36; t += 8) {
        int mt = t >> 2, nt = t & 3;
        float8 acc = {};
#pragma unroll
        for (int kt = 0; kt < 5; ++kt) {
            half16 a = ldsFragA(attnH, mt * 16, 144, kt * 32, 144);
            half16 b = ldsFragB(vT,    nt * 16, 160, kt * 32, 144);
            acc = wmma32(a, b, acc);
        }
        int n  = nt * 16 + (lane & 15);
        int mb = mt * 16 + ((lane >> 4) << 3);
#pragma unroll
        for (int r = 0; r < 8; ++r)
            ctx[base + (size_t)(mb + r) * 64 + n] = acc[r];
    }
}

// ---------------------------------------------------------------------------
// Conv kernel: one workgroup per (b,h); 3x3 conv = 9 shifted WMMA GEMMs, then
// 1x1 conv GEMM; result added into ctx. Weights staged to LDS via the TDM
// (TENSOR_LOAD_TO_LDS, overlapped with the v f32->f16 VALU conversion).
// ---------------------------------------------------------------------------
#define LC_V  0
#define LC_W  55296
#define LC_PW 129024
#define LC_T1 137216
#define LC_SIZE 155648

__global__ __launch_bounds__(256) void conv_kernel(const float* __restrict__ v,
                                                   const _Float16* __restrict__ dwH,
                                                   const _Float16* __restrict__ pwH,
                                                   const float* __restrict__ dwb,
                                                   const float* __restrict__ pwb,
                                                   float* __restrict__ ctx) {
    __shared__ __align__(16) unsigned char smem[LC_SIZE];
    _Float16* v3  = (_Float16*)(smem + LC_V);    // [3][144][64]
    _Float16* wL  = (_Float16*)(smem + LC_W);    // [9][64][64] (tap,o,i)
    _Float16* pwL = (_Float16*)(smem + LC_PW);   // [64][64]    (o,i)
    _Float16* t1h = (_Float16*)(smem + LC_T1);   // [144][64]

    const int tid = threadIdx.x;
    const int lane = tid & 31, wid = tid >> 5;
    const int b = blockIdx.x >> 4, h = blockIdx.x & 15;

#if HAVE_TDM
    if (wid == 0) {                               // wave-level DMA, EXEC ignored
        tdm_copy_f16_1d(wL,  dwH, 36864);
        tdm_copy_f16_1d(pwL, pwH, 4096);
    }
#endif

    for (int i = tid; i < 3 * 9216; i += 256) {
        int hh = i / 9216, rem = i % 9216;
        int hs = h + hh - 1;
        float x = (hs >= 0 && hs < 16) ? v[((size_t)b * 16 + hs) * 9216 + rem] : 0.f;
        v3[i] = (_Float16)x;
    }

#if HAVE_TDM
    if (wid == 0) __builtin_amdgcn_s_wait_tensorcnt(0);
#else
    {   // fallback: f16 weights copied as dwords
        const uint32_t* s0 = (const uint32_t*)dwH; uint32_t* d0 = (uint32_t*)wL;
        for (int i = tid; i < 18432; i += 256) d0[i] = s0[i];
        const uint32_t* s1 = (const uint32_t*)pwH; uint32_t* d1 = (uint32_t*)pwL;
        for (int i = tid; i < 2048; i += 256) d1[i] = s1[i];
    }
#endif
    __syncthreads();

    // ---- 3x3 conv: t1 = sum_tap shift(v) @ w_tap + dw_b --------------------
    for (int t = wid; t < 36; t += 8) {
        int mt = t >> 2, nt = t & 3;
        float8 acc = {};
        for (int tap = 0; tap < 9; ++tap) {
            int dl = (tap % 3) - 1;
            const _Float16* vt = v3 + (tap / 3) * 9216;
            const _Float16* wt = wL + tap * 4096;
#pragma unroll
            for (int kt = 0; kt < 2; ++kt) {
                half16 a  = ldsFragAShift(vt, mt * 16, dl, 64, kt * 32);
                half16 bf = ldsFragB(wt, nt * 16, 64, kt * 32, 64);
                acc = wmma32(a, bf, acc);
            }
        }
        int n  = nt * 16 + (lane & 15);
        int mb = mt * 16 + ((lane >> 4) << 3);
        float bi = dwb[n];
#pragma unroll
        for (int r = 0; r < 8; ++r)
            t1h[(mb + r) * 64 + n] = (_Float16)(acc[r] + bi);
    }
    __syncthreads();

    // ---- 1x1 conv: ctx += t1 @ pw^T + pw_b ---------------------------------
    for (int t = wid; t < 36; t += 8) {
        int mt = t >> 2, nt = t & 3;
        float8 acc = {};
#pragma unroll
        for (int kt = 0; kt < 2; ++kt) {
            acc = wmma32(ldsFragA(t1h, mt * 16, 64, kt * 32, 64),
                         ldsFragB(pwL, nt * 16, 64, kt * 32, 64), acc);
        }
        int n  = nt * 16 + (lane & 15);
        int mb = mt * 16 + ((lane >> 4) << 3);
        float bo = pwb[n];
        size_t obase = (size_t)blockIdx.x * 9216;
#pragma unroll
        for (int r = 0; r < 8; ++r) {
            size_t idx = obase + (size_t)(mb + r) * 64 + n;
            ctx[idx] += acc[r] + bo;
        }
    }
}

// ---------------------------------------------------------------------------
extern "C" void kernel_launch(void* const* d_in, const int* in_sizes, int n_in,
                              void* d_out, int out_size, void* d_ws, size_t ws_size,
                              hipStream_t stream) {
    (void)in_sizes; (void)n_in; (void)out_size; (void)ws_size;
    const float* q    = (const float*)d_in[0];
    const float* k    = (const float*)d_in[1];
    const float* v    = (const float*)d_in[2];
    const float* dw_w = (const float*)d_in[3];
    const float* dw_b = (const float*)d_in[4];
    const float* pw_w = (const float*)d_in[5];
    const float* pw_b = (const float*)d_in[6];
    const float* bias = (const float*)d_in[8];

    float* ctx  = (float*)d_out;                       // (128,16,144,64)
    float* attn = ctx + (size_t)128 * 16 * 144 * 64;   // (128,16,144,144)

    float*    ppTd = (float*)d_ws;
    _Float16* dwH  = (_Float16*)((char*)d_ws + 2048);
    _Float16* pwH  = (_Float16*)((char*)d_ws + 75776);

    prep_kernel<<<161, 256, 0, stream>>>(dw_w, pw_w, ppTd, dwH, pwH);
    attn_kernel<<<2048, 256, 0, stream>>>(q, k, v, ppTd, bias, ctx, attn);
    conv_kernel<<<2048, 256, 0, stream>>>(v, dwH, pwH, dw_b, pw_b, ctx);
}
